// ParticleNet_39917426049315
// MI455X (gfx1250) — matrix operand order repaired and checked by reading, not compiled, not run on hardware
//
#include <hip/hip_runtime.h>
#include <hip/hip_bf16.h>

// ---------------------------------------------------------------------------
// ParticleNet forward for MI455X (gfx1250, wave32).
// - GEMMs on the WMMA pipe: v_wmma_f32_16x16x32_f16 (f16 in, f32 accum).
// - Edge activations stored f16 (halves the dominant HBM traffic, feeds WMMA
//   B-fragments without per-element cvt).
// - BN batch statistics fused into the GEMM epilogue (LDS reduce + one
//   global atomic per channel per block) -> no standalone stats pass.
// - All activations channel-major (C, positions) so GEMM N is contiguous.
// Required ws_size ~ 250 MB (64x524288 f16 + 64x524288 f32 edge buffers).
// ---------------------------------------------------------------------------

typedef __attribute__((ext_vector_type(16))) _Float16 v16h;
typedef __attribute__((ext_vector_type(8)))  float    v8f;

#define BATCH 32
#define NPTS  1024
#define KNN   16
#define BP    (BATCH * NPTS)        // 32768 positions
#define NEDGE (BP * KNN)            // 524288 edges
#define EPSBN 1e-5f

// ---------------------------------------------------------------- utilities
__global__ void zero_kernel(float* p, int n) {
    int i = blockIdx.x * blockDim.x + threadIdx.x;
    if (i < n) p[i] = 0.f;
}

// mask[i] = any(|features[b,:,n]|) != 0
__global__ void mask_kernel(const float* __restrict__ feat, float* __restrict__ maskf) {
    int i = blockIdx.x * blockDim.x + threadIdx.x;
    if (i >= BP) return;
    int b = i / NPTS, n = i % NPTS;
    float s = 0.f;
    #pragma unroll
    for (int c = 0; c < 16; ++c) s += fabsf(feat[(b * 16 + c) * NPTS + n]);
    maskf[i] = (s != 0.f) ? 1.f : 0.f;
}

__global__ void counts_kernel(const float* __restrict__ maskf, float* __restrict__ counts) {
    __shared__ float red[256];
    int b = blockIdx.x;
    float s = 0.f;
    for (int n = threadIdx.x; n < NPTS; n += 256) s += maskf[b * NPTS + n];
    red[threadIdx.x] = s; __syncthreads();
    for (int o = 128; o > 0; o >>= 1) {
        if (threadIdx.x < o) red[threadIdx.x] += red[threadIdx.x + o];
        __syncthreads();
    }
    if (threadIdx.x == 0) counts[b] = fmaxf(red[0], 1.f);
}

// ------------------------------------------------------ input-feature BN
// stats of features*mask over (B,P) per channel; grid = (16, BATCH)
__global__ void bnfts_stats_kernel(const float* __restrict__ feat, const float* __restrict__ maskf,
                                   float* __restrict__ sum, float* __restrict__ sq) {
    int c = blockIdx.x, b = blockIdx.y;
    __shared__ float rs[256], rq[256];
    float s = 0.f, q = 0.f;
    for (int n = threadIdx.x; n < NPTS; n += 256) {
        float v = feat[(b * 16 + c) * NPTS + n] * maskf[b * NPTS + n];
        s += v; q += v * v;
    }
    rs[threadIdx.x] = s; rq[threadIdx.x] = q; __syncthreads();
    for (int o = 128; o > 0; o >>= 1) {
        if (threadIdx.x < o) { rs[threadIdx.x] += rs[threadIdx.x + o]; rq[threadIdx.x] += rq[threadIdx.x + o]; }
        __syncthreads();
    }
    if (threadIdx.x == 0) { atomicAdd(&sum[c], rs[0]); atomicAdd(&sq[c], rq[0]); }
}

__global__ void bn_finalize_kernel(const float* __restrict__ sum, const float* __restrict__ sq,
                                   const float* __restrict__ g, const float* __restrict__ bb,
                                   float invN, int C, float* __restrict__ scale, float* __restrict__ shift) {
    int c = blockIdx.x * blockDim.x + threadIdx.x;
    if (c >= C) return;
    float mean = sum[c] * invN;
    float var  = sq[c] * invN - mean * mean;
    float inv  = rsqrtf(var + EPSBN);
    float sc   = g[c] * inv;
    scale[c] = sc;
    shift[c] = bb[c] - mean * sc;
}

// fts0[c*BP + i] = (feat*mask * scale + shift) * mask   (channel-major)
__global__ void fts0_kernel(const float* __restrict__ feat, const float* __restrict__ maskf,
                            const float* __restrict__ scale, const float* __restrict__ shift,
                            float* __restrict__ fts0) {
    int t = blockIdx.x * blockDim.x + threadIdx.x;
    if (t >= 16 * BP) return;
    int c = t / BP, i = t % BP;
    int b = i / NPTS, n = i % NPTS;
    float m = maskf[i];
    float v = feat[(b * 16 + c) * NPTS + n] * m;
    fts0[t] = (v * scale[c] + shift[c]) * m;
}

// knn coordinates: ec0 uses masked points + coord_shift
__global__ void pknn0_kernel(const float* __restrict__ pts_in, const float* __restrict__ maskf,
                             float* __restrict__ pknn) {
    int t = blockIdx.x * blockDim.x + threadIdx.x;
    if (t >= 3 * BP) return;
    int c = t / BP, i = t % BP;
    int b = i / NPTS, n = i % NPTS;
    float m = maskf[i];
    pknn[t] = pts_in[(b * 3 + c) * NPTS + n] * m + (1.f - m) * 1e9f;
}

// ec1 uses fts (already masked) + coord_shift broadcast over channels
__global__ void pknn1_kernel(const float* __restrict__ fts, const float* __restrict__ maskf,
                             float* __restrict__ pknn) {
    int t = blockIdx.x * blockDim.x + threadIdx.x;
    if (t >= 32 * BP) return;
    int i = t % BP;
    float m = maskf[i];
    pknn[t] = fts[t] + (1.f - m) * 1e9f;
}

// ------------------------------------------------------------------- kNN
// One wave32 per query point; per-lane sorted top-16, merged via LDS.
template <int C>
__global__ void knn_kernel(const float* __restrict__ pts, int* __restrict__ idx) {
    const int lane = threadIdx.x & 31;
    const int wave = threadIdx.x >> 5;
    const int row  = blockIdx.x * 8 + wave;     // [0, BP)
    const int b = row / NPTS, n = row % NPTS;

    __shared__ float sdist[8][32][KNN];
    __shared__ int   sidx [8][32][KNN];
    __shared__ int   shead[8][32];

    float q[C];
    #pragma unroll
    for (int c = 0; c < C; ++c) q[c] = pts[c * BP + b * NPTS + n];

    float bd[KNN]; int bi[KNN];
    #pragma unroll
    for (int t = 0; t < KNN; ++t) { bd[t] = -__builtin_inff(); bi[t] = 0; }

    for (int m = lane; m < NPTS; m += 32) {
        float d2 = 0.f;
        #pragma unroll
        for (int c = 0; c < C; ++c) {
            float df = q[c] - pts[c * BP + b * NPTS + m];
            d2 += df * df;
        }
        float pd = (m == n) ? -__builtin_inff() : -d2;   // self excluded (ref drops rank-0 = self)
        if (pd > bd[KNN - 1]) {
            int t = KNN - 1;
            while (t > 0 && pd > bd[t - 1]) { bd[t] = bd[t - 1]; bi[t] = bi[t - 1]; --t; }
            bd[t] = pd; bi[t] = m;
        }
    }
    #pragma unroll
    for (int t = 0; t < KNN; ++t) { sdist[wave][lane][t] = bd[t]; sidx[wave][lane][t] = bi[t]; }
    shead[wave][lane] = 0;
    __syncthreads();

    if (lane == 0) {  // 32-way merge of sorted lists
        for (int t = 0; t < KNN; ++t) {
            float best = -__builtin_inff(); int bl = 0;
            for (int l = 0; l < 32; ++l) {
                int h = shead[wave][l];
                if (h < KNN && sdist[wave][l][h] > best) { best = sdist[wave][l][h]; bl = l; }
            }
            idx[row * KNN + t] = sidx[wave][bl][shead[wave][bl]];
            shead[wave][bl]++;
        }
    }
}

// ------------------------------------------------- edge gather + frame rot
// vec = Fi * (Fj^T v)  (T = Fi Fj^T never materialized); output (2C, E) f16
template <int C, int NS, int NV>
__global__ void gather_kernel(const float* __restrict__ fts, const int* __restrict__ idx,
                              const float* __restrict__ frames, _Float16* __restrict__ xbuf) {
    int e = blockIdx.x * blockDim.x + threadIdx.x;
    if (e >= NEDGE) return;
    int bp = e / KNN;
    int b  = bp / NPTS;
    int jg = b * NPTS + idx[e];

    float fi[C], fj[C];
    #pragma unroll
    for (int c = 0; c < C; ++c) { fi[c] = fts[c * BP + bp]; fj[c] = fts[c * BP + jg]; }

    float Fi[16], Fj[16];
    #pragma unroll
    for (int t = 0; t < 16; ++t) { Fi[t] = frames[(size_t)bp * 16 + t]; Fj[t] = frames[(size_t)jg * 16 + t]; }

    float nbv[C];
    #pragma unroll
    for (int c = 0; c < NS; ++c) nbv[c] = fj[c];
    #pragma unroll
    for (int v = 0; v < NV; ++v) {
        float w[4];
        #pragma unroll
        for (int cc = 0; cc < 4; ++cc) {
            float s = 0.f;
            #pragma unroll
            for (int b4 = 0; b4 < 4; ++b4) s += Fj[b4 * 4 + cc] * fj[NS + v * 4 + b4];
            w[cc] = s;
        }
        #pragma unroll
        for (int a = 0; a < 4; ++a) {
            float s = 0.f;
            #pragma unroll
            for (int cc = 0; cc < 4; ++cc) s += Fi[a * 4 + cc] * w[cc];
            nbv[NS + v * 4 + a] = s;
        }
    }
    #pragma unroll
    for (int c = 0; c < C; ++c) {
        xbuf[(size_t)c * NEDGE + e]       = (_Float16)fi[c];
        xbuf[(size_t)(C + c) * NEDGE + e] = (_Float16)(nbv[c] - fi[c]);
    }
}

// ----------------------------------------------------------- WMMA GEMM
// Y[m, e] = sum_k W[m, k] * X[k, e];  Cout = MT*16, K padded to KT*32.
// Block = 128 threads = 4 waves; each wave owns one 16-wide N tile.
// BN batch statistics (per-channel sum / sumsq of Y) fused into the epilogue:
// accumulator values are live in registers -> LDS reduce -> 1 atomic/ch/block.
template <int MT, int KT, int CIN, typename TX>
__global__ void __launch_bounds__(128) wmma_gemm_kernel(const float* __restrict__ W,
                                                        const TX* __restrict__ X,
                                                        float* __restrict__ Y, int N,
                                                        float* __restrict__ ssum,
                                                        float* __restrict__ ssq) {
    __shared__ _Float16 wlds[MT * 16 * KT * 32];
    __shared__ float lsum[MT * 16];
    __shared__ float lsq [MT * 16];
    const int tid = threadIdx.x;
    for (int t = tid; t < MT * 16 * KT * 32; t += 128) {
        int r = t / (KT * 32), k = t % (KT * 32);
        wlds[t] = (k < CIN) ? (_Float16)W[r * CIN + k] : (_Float16)0.f;
    }
    for (int t = tid; t < MT * 16; t += 128) { lsum[t] = 0.f; lsq[t] = 0.f; }
    __syncthreads();

    const int lane = tid & 31, wave = tid >> 5;
    const int n0   = blockIdx.x * 64 + wave * 16;
    const int nl   = n0 + (lane & 15);
    const int h    = lane >> 4;                 // lane half selects K sub-block

    v8f acc[MT];
    v8f zero = {};
    #pragma unroll
    for (int mt = 0; mt < MT; ++mt) acc[mt] = zero;

    #pragma unroll
    for (int kt = 0; kt < KT; ++kt) {
        // B fragment (32x16 f16): lanes 0-15 hold K=0..15, lanes 16-31 K=16..31
        v16h bfrag;
        #pragma unroll
        for (int i = 0; i < 16; ++i) {
            int k = kt * 32 + i + 16 * h;
            bfrag[i] = (k < CIN) ? (_Float16)X[(size_t)k * N + nl] : (_Float16)0.f;
        }
        #pragma unroll
        for (int mt = 0; mt < MT; ++mt) {
            // A fragment (16x32 f16): lanes 0-15 hold K=0..15, lanes 16-31 K=16..31
            v16h afrag;
            #pragma unroll
            for (int i = 0; i < 16; ++i) {
                int koff = i + 16 * h;
                afrag[i] = wlds[(mt * 16 + (lane & 15)) * (KT * 32) + kt * 32 + koff];
            }
            acc[mt] = __builtin_amdgcn_wmma_f32_16x16x32_f16(
                false, afrag, false, bfrag, (short)0, acc[mt], false, false);
        }
    }
    // D layout: lane<16 -> rows v+0..7, lane>=16 -> rows v+8..15, col = lane&15
    #pragma unroll
    for (int mt = 0; mt < MT; ++mt) {
        #pragma unroll
        for (int v = 0; v < 8; ++v) {
            int m = mt * 16 + v + 8 * h;
            float val = acc[mt][v];
            Y[(size_t)m * N + nl] = val;
            atomicAdd(&lsum[m], val);
            atomicAdd(&lsq[m],  val * val);
        }
    }
    __syncthreads();
    for (int t = tid; t < MT * 16; t += 128) {
        atomicAdd(&ssum[t], lsum[t]);
        atomicAdd(&ssq[t],  lsq[t]);
    }
}

// out = relu(y*scale+shift [+ add]) [* mask];  tensors are (C, N) channel-major
template <typename TO>
__global__ void bn_relu_kernel(const float* __restrict__ y, const float* __restrict__ scale,
                               const float* __restrict__ shift, const float* __restrict__ add,
                               const float* __restrict__ maskf, int C, int N, TO* __restrict__ out) {
    size_t t = (size_t)blockIdx.x * blockDim.x + threadIdx.x;
    if (t >= (size_t)C * N) return;
    int c = (int)(t / N);
    size_t r = t % N;
    float v = y[t] * scale[c] + shift[c];
    if (add)   v += add[t];
    v = fmaxf(v, 0.f);
    if (maskf) v *= maskf[r];
    out[t] = (TO)v;
}

// agg[c, bp] = mean_k x[c, bp*K + k]  (each thread reads one contiguous 32B run)
__global__ void aggregate_kernel(const _Float16* __restrict__ x, int C, float* __restrict__ agg) {
    size_t t = (size_t)blockIdx.x * blockDim.x + threadIdx.x;
    if (t >= (size_t)C * BP) return;
    int c = (int)(t / BP);
    size_t bp = t % BP;
    const _Float16* p = x + (size_t)c * NEDGE + bp * KNN;
    float s = 0.f;
    #pragma unroll
    for (int k = 0; k < KNN; ++k) s += (float)p[k];
    agg[t] = s * (1.f / KNN);
}

// pooled[b, c] = sum_n fused[c, b*P+n] / counts[b];  grid = BATCH*128
__global__ void pool_kernel(const float* __restrict__ fused, const float* __restrict__ counts,
                            float* __restrict__ pooled) {
    int b = blockIdx.x / 128, c = blockIdx.x % 128;
    __shared__ float red[256];
    float s = 0.f;
    for (int n = threadIdx.x; n < NPTS; n += 256) s += fused[(size_t)c * BP + b * NPTS + n];
    red[threadIdx.x] = s; __syncthreads();
    for (int o = 128; o > 0; o >>= 1) {
        if (threadIdx.x < o) red[threadIdx.x] += red[threadIdx.x + o];
        __syncthreads();
    }
    if (threadIdx.x == 0) pooled[b * 128 + c] = red[0] / counts[b];
}

__global__ void fc1_kernel(const float* __restrict__ x, const float* __restrict__ w,
                           const float* __restrict__ bias, float* __restrict__ out) {
    int b = blockIdx.x, o = threadIdx.x;   // 128 threads
    float s = bias[o];
    #pragma unroll 4
    for (int c = 0; c < 128; ++c) s += x[b * 128 + c] * w[o * 128 + c];
    out[b * 128 + o] = fmaxf(s, 0.f);
}

__global__ void fc2_kernel(const float* __restrict__ x, const float* __restrict__ w,
                           const float* __restrict__ bias, float* __restrict__ out) {
    int t = blockIdx.x * blockDim.x + threadIdx.x;
    if (t >= BATCH * 10) return;
    int b = t / 10, o = t % 10;
    float s = bias[o];
    #pragma unroll 4
    for (int c = 0; c < 128; ++c) s += x[b * 128 + c] * w[o * 128 + c];
    out[t] = s;
}

// ---------------------------------------------------------------------------
extern "C" void kernel_launch(void* const* d_in, const int* in_sizes, int n_in,
                              void* d_out, int out_size, void* d_ws, size_t ws_size,
                              hipStream_t stream) {
    // jax pytree flatten order (dicts sorted by key):
    const float* points   = (const float*)d_in[0];
    const float* features = (const float*)d_in[1];
    const float* frames   = (const float*)d_in[2];
    const float* bnfts_b  = (const float*)d_in[3];
    const float* bnfts_g  = (const float*)d_in[4];
    const float* e0c_b[3] = {(const float*)d_in[5],  (const float*)d_in[8],  (const float*)d_in[11]};
    const float* e0c_g[3] = {(const float*)d_in[6],  (const float*)d_in[9],  (const float*)d_in[12]};
    const float* e0c_w[3] = {(const float*)d_in[7],  (const float*)d_in[10], (const float*)d_in[13]};
    const float* e0_scb = (const float*)d_in[14];
    const float* e0_scg = (const float*)d_in[15];
    const float* e0_scw = (const float*)d_in[16];
    const float* e1c_b[3] = {(const float*)d_in[17], (const float*)d_in[20], (const float*)d_in[23]};
    const float* e1c_g[3] = {(const float*)d_in[18], (const float*)d_in[21], (const float*)d_in[24]};
    const float* e1c_w[3] = {(const float*)d_in[19], (const float*)d_in[22], (const float*)d_in[25]};
    const float* e1_scb = (const float*)d_in[26];
    const float* e1_scg = (const float*)d_in[27];
    const float* e1_scw = (const float*)d_in[28];
    const float* fc1_b  = (const float*)d_in[29];
    const float* fc1_w  = (const float*)d_in[30];
    const float* fc2_b  = (const float*)d_in[31];
    const float* fc2_w  = (const float*)d_in[32];
    const float* fus_b  = (const float*)d_in[33];
    const float* fus_g  = (const float*)d_in[34];
    const float* fus_w  = (const float*)d_in[35];

    // ---- workspace carve-up ----
    float* p = (float*)d_ws;
    float* maskf  = p; p += BP;
    float* counts = p; p += 32;
    float* ssum   = p; p += 128;     // ssum/ssq adjacent -> one zero of 256
    float* ssq    = p; p += 128;
    float* scale  = p; p += 128;
    float* shift  = p; p += 128;
    float* fts0   = p; p += (size_t)16 * BP;
    float* out0   = p; p += (size_t)32 * BP;    // ec0 output; fusion input ch 0..31
    float* out1   = p; p += (size_t)64 * BP;    // ec1 output; contiguous -> free concat
    float* pknn   = p; p += (size_t)32 * BP;
    float* scb    = p; p += (size_t)64 * BP;    // shortcut GEMM output
    float* agg    = p; p += (size_t)64 * BP;
    float* fused  = p; p += (size_t)128 * BP;
    float* pooled = p; p += 32 * 128;
    float* fc1o   = p; p += 32 * 128;
    int*   idxb   = (int*)p; p += NEDGE;
    float* ybuf   = p; p += (size_t)64 * NEDGE;         // f32 GEMM outputs
    _Float16* xbuf = (_Float16*)p;                      // f16 edge activations (64 x NEDGE)

    const float invBP = 1.f / (float)BP;
    const float invE  = 1.f / (float)NEDGE;

    // ---- prologue: mask, counts, input BN ----
    mask_kernel<<<(BP + 255) / 256, 256, 0, stream>>>(features, maskf);
    counts_kernel<<<BATCH, 256, 0, stream>>>(maskf, counts);
    zero_kernel<<<1, 256, 0, stream>>>(ssum, 256);
    bnfts_stats_kernel<<<dim3(16, BATCH), 256, 0, stream>>>(features, maskf, ssum, ssq);
    bn_finalize_kernel<<<1, 128, 0, stream>>>(ssum, ssq, bnfts_g, bnfts_b, invBP, 16, scale, shift);
    fts0_kernel<<<(16 * BP + 255) / 256, 256, 0, stream>>>(features, maskf, scale, shift, fts0);

    // =============================== EdgeConv 0 (Cin=16, Cout=32, ns=4, nv=3)
    pknn0_kernel<<<(3 * BP + 255) / 256, 256, 0, stream>>>(points, maskf, pknn);
    knn_kernel<3><<<BP / 8, 256, 0, stream>>>(pknn, idxb);
    gather_kernel<16, 4, 3><<<(NEDGE + 255) / 256, 256, 0, stream>>>(fts0, idxb, frames, xbuf);
    for (int l = 0; l < 3; ++l) {   // 32x32 convs over E edges
        zero_kernel<<<1, 256, 0, stream>>>(ssum, 256);
        wmma_gemm_kernel<2, 1, 32, _Float16><<<NEDGE / 64, 128, 0, stream>>>(
            e0c_w[l], xbuf, ybuf, NEDGE, ssum, ssq);
        bn_finalize_kernel<<<1, 128, 0, stream>>>(ssum, ssq, e0c_g[l], e0c_b[l], invE, 32, scale, shift);
        bn_relu_kernel<_Float16><<<(int)(((size_t)32 * NEDGE + 255) / 256), 256, 0, stream>>>(
            ybuf, scale, shift, nullptr, nullptr, 32, NEDGE, xbuf);
    }
    aggregate_kernel<<<(32 * BP + 255) / 256, 256, 0, stream>>>(xbuf, 32, agg);
    zero_kernel<<<1, 256, 0, stream>>>(ssum, 256);
    wmma_gemm_kernel<2, 1, 16, float><<<BP / 64, 128, 0, stream>>>(
        e0_scw, fts0, scb, BP, ssum, ssq);
    bn_finalize_kernel<<<1, 128, 0, stream>>>(ssum, ssq, e0_scg, e0_scb, invBP, 32, scale, shift);
    bn_relu_kernel<float><<<(32 * BP + 255) / 256, 256, 0, stream>>>(
        scb, scale, shift, agg, maskf, 32, BP, out0);

    // =============================== EdgeConv 1 (Cin=32, Cout=64, ns=8, nv=6)
    pknn1_kernel<<<(32 * BP + 255) / 256, 256, 0, stream>>>(out0, maskf, pknn);
    knn_kernel<32><<<BP / 8, 256, 0, stream>>>(pknn, idxb);
    gather_kernel<32, 8, 6><<<(NEDGE + 255) / 256, 256, 0, stream>>>(out0, idxb, frames, xbuf);
    for (int l = 0; l < 3; ++l) {   // 64x64 convs over E edges
        zero_kernel<<<1, 256, 0, stream>>>(ssum, 256);
        wmma_gemm_kernel<4, 2, 64, _Float16><<<NEDGE / 64, 128, 0, stream>>>(
            e1c_w[l], xbuf, ybuf, NEDGE, ssum, ssq);
        bn_finalize_kernel<<<1, 128, 0, stream>>>(ssum, ssq, e1c_g[l], e1c_b[l], invE, 64, scale, shift);
        bn_relu_kernel<_Float16><<<(int)(((size_t)64 * NEDGE + 255) / 256), 256, 0, stream>>>(
            ybuf, scale, shift, nullptr, nullptr, 64, NEDGE, xbuf);
    }
    aggregate_kernel<<<(64 * BP + 255) / 256, 256, 0, stream>>>(xbuf, 64, agg);
    zero_kernel<<<1, 256, 0, stream>>>(ssum, 256);
    wmma_gemm_kernel<4, 1, 32, float><<<BP / 64, 128, 0, stream>>>(
        e1_scw, out0, scb, BP, ssum, ssq);
    bn_finalize_kernel<<<1, 128, 0, stream>>>(ssum, ssq, e1_scg, e1_scb, invBP, 64, scale, shift);
    bn_relu_kernel<float><<<(64 * BP + 255) / 256, 256, 0, stream>>>(
        scb, scale, shift, agg, maskf, 64, BP, out1);

    // =============================== Fusion (128 x 96); concat is free: out0||out1
    zero_kernel<<<1, 256, 0, stream>>>(ssum, 256);
    wmma_gemm_kernel<8, 3, 96, float><<<BP / 64, 128, 0, stream>>>(
        fus_w, out0, ybuf, BP, ssum, ssq);
    bn_finalize_kernel<<<1, 128, 0, stream>>>(ssum, ssq, fus_g, fus_b, invBP, 128, scale, shift);
    bn_relu_kernel<float><<<(128 * BP + 255) / 256, 256, 0, stream>>>(
        ybuf, scale, shift, nullptr, maskf, 128, BP, fused);

    // =============================== Head
    pool_kernel<<<BATCH * 128, 256, 0, stream>>>(fused, counts, pooled);
    fc1_kernel<<<BATCH, 128, 0, stream>>>(pooled, fc1_w, fc1_b, fc1o);
    fc2_kernel<<<(BATCH * 10 + 255) / 256, 256, 0, stream>>>(fc1o, fc2_w, fc2_b, (float*)d_out);
}
